// RNNModel_90117003805131
// MI455X (gfx1250) — matrix operand order repaired and checked by reading, not compiled, not run on hardware
//
#include <hip/hip_runtime.h>
#include <math.h>

// ---------------- types for WMMA ----------------
typedef __bf16 bf16_t;
typedef __attribute__((ext_vector_type(16))) __bf16 v16bf;
typedef __attribute__((ext_vector_type(8)))  float  v8f;

struct Chunk16 { unsigned long long x, y; };   // 16 bytes, trivially copyable

// Model dimensions
#define NTOK  33278
#define NHID  1024
#define SEQ   128
#define BATCH 32
#define SB    (SEQ*BATCH)        // 4096
#define G4    (4*NHID)           // 4096

static __device__ __forceinline__ v8f zero8() {
    v8f v = {0.f,0.f,0.f,0.f,0.f,0.f,0.f,0.f};
    return v;
}

// A-fragment (16x32 bf16, split-K layout): p = row base + 8*(lane>>4)
// elems 0..7 <- p[0..7], elems 8..15 <- p[16..23]
static __device__ __forceinline__ v16bf load_fragA_bf16(const bf16_t* p) {
    union { Chunk16 c[2]; v16bf v; } u;
    u.c[0] = *(const Chunk16*)(p);
    u.c[1] = *(const Chunk16*)(p + 16);
    return u.v;
}

static __device__ __forceinline__ v16bf load_fragA_f32(const float* p) {
    v16bf v;
#pragma unroll
    for (int e = 0; e < 8; ++e) v[e]     = (__bf16)p[e];
#pragma unroll
    for (int e = 0; e < 8; ++e) v[e + 8] = (__bf16)p[e + 16];
    return v;
}

// B-fragment (32x16 bf16, split-K): elems 0..7 <- p[0..7], elems 8..15 <- p[16..23] from LDS tile.
// Tile layout: [rows][32 K] bf16 row-major. lane column = localRow, K off = boff.
static __device__ __forceinline__ v16bf ldsB_frag(const bf16_t* tile, int localRow, int boff) {
    const bf16_t* p = tile + localRow * 32 + boff;
    union { Chunk16 c[2]; v16bf v; } u;
    u.c[0] = *(const Chunk16*)(p);
    u.c[1] = *(const Chunk16*)(p + 16);
    return u.v;
}

// Async copy of a [ROWS x 32] bf16 tile (row-major, 64B rows) global -> LDS.
// Each of 256 threads copies 16B chunks; ROWS/64 chunks per thread.
// Tracked on ASYNCcnt; caller pairs with s_wait_asynccnt + barrier.
template <int ROWS>
static __device__ __forceinline__ void async_copy_Btile(const bf16_t* __restrict__ W,
                                                        int n0, int kb,
                                                        unsigned ldsBase, int tid, int nmax) {
#pragma unroll
    for (int j = 0; j < ROWS / 64; ++j) {
        int row   = j * 64 + (tid >> 2);
        int chunk = tid & 3;
        int n = n0 + row; if (n > nmax) n = nmax;
        unsigned long long src =
            (unsigned long long)(const void*)(W + (size_t)n * NHID + kb + chunk * 8);
        unsigned dst = ldsBase + (unsigned)(row * 64 + chunk * 16);
        asm volatile("global_load_async_to_lds_b128 %0, %1, off"
                     :: "v"(dst), "v"(src) : "memory");
    }
}

static __device__ __forceinline__ void wait_async_all() {
    asm volatile("s_wait_asynccnt 0x0" ::: "memory");
}

// ---------------- elementwise helpers ----------------
__global__ void k_f32_to_bf16(const float* __restrict__ in, bf16_t* __restrict__ out, int n) {
    int i = blockIdx.x * blockDim.x + threadIdx.x;
    if (i < n) out[i] = (__bf16)in[i];
}

// ---------------- GEMM 1: pre = emb[tokens] @ w_ih^T + b_ih ----------------
// Block: 8 waves -> 256M x 64N tile (each wave 32M x 64N).
// B staged in LDS (async, double-buffered). Grid: (64 Ntiles, 16 Mtiles).
__global__ void k_gemm_pre(const int* __restrict__ tokens,
                           const float* __restrict__ emb,        // [NTOK][1024] f32
                           const bf16_t* __restrict__ Wih,       // [4096][1024] bf16
                           const float* __restrict__ b_ih,       // [4096]
                           float* __restrict__ pre)              // [4096][4096]
{
    __shared__ bf16_t smem[2][64 * 32];                          // 2 x 4KB
    const int tid  = threadIdx.x;
    const int lane = tid & 31;
    const int wave = tid >> 5;
    const int m0 = blockIdx.y * 256 + wave * 32;                 // 2 M-subtiles
    const int n0 = blockIdx.x * 64;
    const int r    = lane & 15;
    const int aoff = (lane >> 4) << 3;
    const int boff = (lane >> 4) << 3;

    const int tok0 = tokens[m0 + r];
    const int tok1 = tokens[m0 + 16 + r];
    const float* aRow0 = emb + (size_t)tok0 * NHID + aoff;
    const float* aRow1 = emb + (size_t)tok1 * NHID + aoff;

    unsigned sb0 = (unsigned)(unsigned long long)(const void*)&smem[0][0];
    unsigned sb1 = (unsigned)(unsigned long long)(const void*)&smem[1][0];

    async_copy_Btile<64>(Wih, n0, 0, sb0, tid, G4 - 1);
    v16bf aCur0 = load_fragA_f32(aRow0);
    v16bf aCur1 = load_fragA_f32(aRow1);

    v8f acc[8] = { zero8(), zero8(), zero8(), zero8(),
                   zero8(), zero8(), zero8(), zero8() };
    for (int i = 0; i < NHID / 32; ++i) {
        const int kb = i << 5;
        wait_async_all();
        __syncthreads();
        if (i + 1 < NHID / 32)
            async_copy_Btile<64>(Wih, n0, kb + 32, (i & 1) ? sb0 : sb1, tid, G4 - 1);
        v16bf aNxt0 = aCur0, aNxt1 = aCur1;
        if (i + 1 < NHID / 32) {
            aNxt0 = load_fragA_f32(aRow0 + kb + 32);
            aNxt1 = load_fragA_f32(aRow1 + kb + 32);
        }
        const bf16_t* bs = smem[i & 1];
        v16bf bf[4];
#pragma unroll
        for (int s = 0; s < 4; ++s) bf[s] = ldsB_frag(bs, s * 16 + r, boff);
#pragma unroll
        for (int s = 0; s < 4; ++s)
            acc[s] = __builtin_amdgcn_wmma_f32_16x16x32_bf16(
                false, aCur0, false, bf[s], (short)0, acc[s], false, false);
#pragma unroll
        for (int s = 0; s < 4; ++s)
            acc[4 + s] = __builtin_amdgcn_wmma_f32_16x16x32_bf16(
                false, aCur1, false, bf[s], (short)0, acc[4 + s], false, false);
        aCur0 = aNxt0; aCur1 = aNxt1;
    }
    const int mrow = (lane >> 4) << 3;
#pragma unroll
    for (int half = 0; half < 2; ++half) {
#pragma unroll
        for (int s = 0; s < 4; ++s) {
            int col = n0 + s * 16 + (lane & 15);
            float bv = b_ih[col];
#pragma unroll
            for (int v = 0; v < 8; ++v) {
                int m = m0 + half * 16 + mrow + v;
                pre[(size_t)m * G4 + col] = acc[half * 4 + s][v] + bv;
            }
        }
    }
}

// ---------------- GEMM 2 (per timestep): gates = pre_t + h @ w_hh^T + b_hh ----------------
// Block: 8 waves -> 32M x 256N tile (wave w: mt = w&1, ngroup = w>>1).
// B tile 256x32 bf16 staged in LDS (async, double-buffered). Grid: 16 blocks.
__global__ void k_gemm_gates(const float* __restrict__ pre_t,    // [32][4096]
                             const bf16_t* __restrict__ hprev,   // [32][1024] bf16
                             const bf16_t* __restrict__ Whh,     // [4096][1024] bf16
                             const float* __restrict__ b_hh,     // [4096]
                             float* __restrict__ gates)          // [32][4096]
{
    __shared__ bf16_t smem[2][256 * 32];                         // 2 x 16KB
    const int tid  = threadIdx.x;
    const int lane = tid & 31;
    const int wave = tid >> 5;
    const int mt = wave & 1;
    const int ng = wave >> 1;                                    // 0..3
    const int m0 = mt * 16;
    const int nblk = blockIdx.x * 256;
    const int n0 = nblk + ng * 64;
    const int r    = lane & 15;
    const int aoff = (lane >> 4) << 3;
    const int boff = (lane >> 4) << 3;

    const bf16_t* aRow = hprev + (size_t)(m0 + r) * NHID + aoff;

    unsigned sb0 = (unsigned)(unsigned long long)(const void*)&smem[0][0];
    unsigned sb1 = (unsigned)(unsigned long long)(const void*)&smem[1][0];

    async_copy_Btile<256>(Whh, nblk, 0, sb0, tid, G4 - 1);
    v16bf aCur = load_fragA_bf16(aRow);

    v8f acc[4] = { zero8(), zero8(), zero8(), zero8() };
    for (int i = 0; i < NHID / 32; ++i) {
        const int kb = i << 5;
        wait_async_all();
        __syncthreads();
        if (i + 1 < NHID / 32)
            async_copy_Btile<256>(Whh, nblk, kb + 32, (i & 1) ? sb0 : sb1, tid, G4 - 1);
        v16bf aNxt = aCur;
        if (i + 1 < NHID / 32) aNxt = load_fragA_bf16(aRow + kb + 32);
        const bf16_t* bs = smem[i & 1];
        v16bf bf[4];
#pragma unroll
        for (int s = 0; s < 4; ++s) bf[s] = ldsB_frag(bs, ng * 64 + s * 16 + r, boff);
#pragma unroll
        for (int s = 0; s < 4; ++s)
            acc[s] = __builtin_amdgcn_wmma_f32_16x16x32_bf16(
                false, aCur, false, bf[s], (short)0, acc[s], false, false);
        aCur = aNxt;
    }
    const int mrow = (lane >> 4) << 3;
#pragma unroll
    for (int s = 0; s < 4; ++s) {
        int col = n0 + s * 16 + (lane & 15);
        float bv = b_hh[col];
#pragma unroll
        for (int v = 0; v < 8; ++v) {
            int m = m0 + mrow + v;
            gates[(size_t)m * G4 + col] = acc[s][v] + pre_t[(size_t)m * G4 + col] + bv;
        }
    }
}

// ---------------- LSTM pointwise cell update ----------------
__global__ void k_lstm_cell(const float* __restrict__ gates,     // [32][4096]
                            float* __restrict__ c,               // [32][1024] f32 (in/out)
                            bf16_t* __restrict__ h_bf,           // [32][1024] bf16 (= out_all[t])
                            float* __restrict__ h_f32)           // [32][1024] f32
{
    int idx = blockIdx.x * blockDim.x + threadIdx.x;             // 0..32767
    int b = idx >> 10, j = idx & 1023;
    const float* g = gates + (size_t)b * G4;
    float ig = 1.f / (1.f + __expf(-g[j]));
    float fg = 1.f / (1.f + __expf(-g[NHID + j]));
    float gg = tanhf(g[2 * NHID + j]);
    float og = 1.f / (1.f + __expf(-g[3 * NHID + j]));
    float cn = fg * c[idx] + ig * gg;
    c[idx] = cn;
    float h = og * tanhf(cn);
    h_f32[idx] = h;
    h_bf[idx]  = (__bf16)h;
}

// ---------------- GEMM 3: decoded = out_all @ dec_w^T + dec_b ----------------
// Block: 8 waves -> 256M x 64N tile (each wave 32M x 64N).
// B staged in LDS (async, double-buffered). Grid: (520 Ntiles, 16 Mtiles).
__global__ void k_gemm_dec(const bf16_t* __restrict__ A,         // [4096][1024] bf16
                           const bf16_t* __restrict__ W,         // [33278][1024] bf16
                           const float* __restrict__ bias,       // [33278]
                           float* __restrict__ C)                // [4096][33278]
{
    __shared__ bf16_t smem[2][64 * 32];                          // 2 x 4KB
    const int tid  = threadIdx.x;
    const int lane = tid & 31;
    const int wave = tid >> 5;
    const int m0 = blockIdx.y * 256 + wave * 32;                 // 2 M-subtiles
    const int n0 = blockIdx.x * 64;
    const int r    = lane & 15;
    const int aoff = (lane >> 4) << 3;
    const int boff = (lane >> 4) << 3;

    const bf16_t* aRow0 = A + (size_t)(m0 + r) * NHID + aoff;
    const bf16_t* aRow1 = A + (size_t)(m0 + 16 + r) * NHID + aoff;

    unsigned sb0 = (unsigned)(unsigned long long)(const void*)&smem[0][0];
    unsigned sb1 = (unsigned)(unsigned long long)(const void*)&smem[1][0];

    async_copy_Btile<64>(W, n0, 0, sb0, tid, NTOK - 1);
    v16bf aCur0 = load_fragA_bf16(aRow0);
    v16bf aCur1 = load_fragA_bf16(aRow1);

    v8f acc[8] = { zero8(), zero8(), zero8(), zero8(),
                   zero8(), zero8(), zero8(), zero8() };
    for (int i = 0; i < NHID / 32; ++i) {
        const int kb = i << 5;
        wait_async_all();
        __syncthreads();
        if (i + 1 < NHID / 32)
            async_copy_Btile<64>(W, n0, kb + 32, (i & 1) ? sb0 : sb1, tid, NTOK - 1);
        v16bf aNxt0 = aCur0, aNxt1 = aCur1;
        if (i + 1 < NHID / 32) {
            __builtin_prefetch(aRow0 + kb + 64, 0, 3);
            aNxt0 = load_fragA_bf16(aRow0 + kb + 32);
            aNxt1 = load_fragA_bf16(aRow1 + kb + 32);
        }
        const bf16_t* bs = smem[i & 1];
        v16bf bf[4];
#pragma unroll
        for (int s = 0; s < 4; ++s) bf[s] = ldsB_frag(bs, s * 16 + r, boff);
#pragma unroll
        for (int s = 0; s < 4; ++s)
            acc[s] = __builtin_amdgcn_wmma_f32_16x16x32_bf16(
                false, aCur0, false, bf[s], (short)0, acc[s], false, false);
#pragma unroll
        for (int s = 0; s < 4; ++s)
            acc[4 + s] = __builtin_amdgcn_wmma_f32_16x16x32_bf16(
                false, aCur1, false, bf[s], (short)0, acc[4 + s], false, false);
        aCur0 = aNxt0; aCur1 = aNxt1;
    }
    const int mrow = (lane >> 4) << 3;
#pragma unroll
    for (int half = 0; half < 2; ++half) {
#pragma unroll
        for (int s = 0; s < 4; ++s) {
            int col = n0 + s * 16 + (lane & 15);
            if (col < NTOK) {
                float bv = bias[col];
#pragma unroll
                for (int v = 0; v < 8; ++v) {
                    int m = m0 + half * 16 + mrow + v;
                    C[(size_t)m * NTOK + col] = acc[half * 4 + s][v] + bv;
                }
            }
        }
    }
}

// ---------------- tail: append hT, cT to output ----------------
__global__ void k_tail(const float* __restrict__ h, const float* __restrict__ c,
                       float* __restrict__ out) {
    int i = blockIdx.x * blockDim.x + threadIdx.x;
    if (i < BATCH * NHID) {
        out[i] = h[i];
        out[BATCH * NHID + i] = c[i];
    }
}

// ---------------- host-side orchestration ----------------
extern "C" void kernel_launch(void* const* d_in, const int* in_sizes, int n_in,
                              void* d_out, int out_size, void* d_ws, size_t ws_size,
                              hipStream_t stream) {
    const int*   tokens = (const int*)  d_in[0];   // [128,32]
    const float* h0     = (const float*)d_in[1];   // [1,32,1024]
    const float* c0     = (const float*)d_in[2];
    const float* emb    = (const float*)d_in[3];   // [33278,1024]
    const float* w_ih   = (const float*)d_in[4];   // [4096,1024]
    const float* w_hh   = (const float*)d_in[5];   // [4096,1024]
    const float* b_ih   = (const float*)d_in[6];
    const float* b_hh   = (const float*)d_in[7];
    const float* dec_w  = (const float*)d_in[8];   // [33278,1024]
    const float* dec_b  = (const float*)d_in[9];

    float* out = (float*)d_out;                    // decoded (4096*33278) ++ hT ++ cT

    // workspace bump allocator (256B aligned)
    char* wsp = (char*)d_ws;
    auto alloc = [&](size_t bytes) -> void* {
        void* p = (void*)wsp;
        wsp += (bytes + 255) & ~(size_t)255;
        return p;
    };
    bf16_t* wih_bf  = (bf16_t*)alloc((size_t)G4 * NHID * 2);       // 8 MB
    bf16_t* whh_bf  = (bf16_t*)alloc((size_t)G4 * NHID * 2);       // 8 MB
    bf16_t* decw_bf = (bf16_t*)alloc((size_t)NTOK * NHID * 2);     // 68 MB
    float*  pre     = (float*) alloc((size_t)SB * G4 * 4);         // 64 MB
    bf16_t* out_all = (bf16_t*)alloc((size_t)SB * NHID * 2);       // 8 MB
    bf16_t* hzero   = (bf16_t*)alloc((size_t)BATCH * NHID * 2);
    float*  cbuf    = (float*) alloc((size_t)BATCH * NHID * 4);
    float*  hf32    = (float*) alloc((size_t)BATCH * NHID * 4);
    float*  gates   = (float*) alloc((size_t)BATCH * G4 * 4);

    // 1) weight conversions f32 -> bf16
    {
        int n;
        n = G4 * NHID;
        k_f32_to_bf16<<<(n + 255) / 256, 256, 0, stream>>>(w_ih, wih_bf, n);
        k_f32_to_bf16<<<(n + 255) / 256, 256, 0, stream>>>(w_hh, whh_bf, n);
        n = NTOK * NHID;
        k_f32_to_bf16<<<(n + 255) / 256, 256, 0, stream>>>(dec_w, decw_bf, n);
        n = BATCH * NHID;
        k_f32_to_bf16<<<(n + 255) / 256, 256, 0, stream>>>(h0, hzero, n);
    }
    // c = c0
    hipMemcpyAsync(cbuf, c0, (size_t)BATCH * NHID * 4, hipMemcpyDeviceToDevice, stream);

    // 2) pre = emb[tokens] @ w_ih^T + b_ih
    k_gemm_pre<<<dim3(64, 16), 256, 0, stream>>>(tokens, emb, wih_bf, b_ih, pre);

    // 3) LSTM scan: 128 sequential steps
    for (int t = 0; t < SEQ; ++t) {
        const bf16_t* hprev = (t == 0) ? hzero : (out_all + (size_t)(t - 1) * BATCH * NHID);
        const float*  pre_t = pre + (size_t)t * BATCH * G4;
        bf16_t* h_bf = out_all + (size_t)t * BATCH * NHID;
        k_gemm_gates<<<16, 256, 0, stream>>>(pre_t, hprev, whh_bf, b_hh, gates);
        k_lstm_cell<<<(BATCH * NHID) / 256, 256, 0, stream>>>(gates, cbuf, h_bf, hf32);
    }

    // 4) decoder GEMM: C = out_all @ dec_w^T + dec_b
    k_gemm_dec<<<dim3(520, 16), 256, 0, stream>>>(out_all, decw_bf, dec_b, out);

    // 5) append hT, cT
    k_tail<<<(BATCH * NHID + 255) / 256, 256, 0, stream>>>(
        hf32, cbuf, out + (size_t)SB * NTOK);
}
